// EncodeProcessDecodeNetwork_7335804142387
// MI455X (gfx1250) — matrix operand order repaired and checked by reading, not compiled, not run on hardware
//
#include <hip/hip_runtime.h>

// ---------------------------------------------------------------------------
// CDNA5 (gfx1250) fused GNN: WMMA f32<-f16 16x16x32 for every MLP layer.
// ---------------------------------------------------------------------------

typedef _Float16 v16h __attribute__((ext_vector_type(16)));
typedef _Float16 v8h  __attribute__((ext_vector_type(8)));
typedef float    v8f  __attribute__((ext_vector_type(8)));

struct MlpP {
  const float* bi;     // bias layer1 [64]
  const float* bh;     // bias hidden [64]
  const float* bo;     // bias out    [outn]
  const float* gamma;  // [32] or null
  const float* beta;   // [32] or null
  const v16h*  p1;     // packed W1 tiles: ks1*4 tiles
  const v16h*  p2;     // packed W2 tiles: 2*4 tiles
  const v16h*  p3;     // packed W3 tiles: 2*cts3 tiles
};

__device__ __forceinline__ v8f wmma32(v16h a, v16h b, v8f c) {
  // (neg_a, A, neg_b, B, c_mod, C, reuse_a, reuse_b)
  return __builtin_amdgcn_wmma_f32_16x16x32_f16(false, a, false, b, (short)0, c,
                                                false, false);
}

__device__ __forceinline__ v8f zero8() {
  v8f z;
#pragma unroll
  for (int i = 0; i < 8; ++i) z[i] = 0.f;
  return z;
}

// A-layout K permutation (16-bit A, 16x32): lane half h, slot i -> k in [0,32)
__device__ __forceinline__ int aperm(int h, int i) {
  return (i < 8) ? (i + 8 * h) : (i + 8 + 8 * h);
}

// Build one A k-step (16 rows x 32 k) from a 32-wide f32 row in global memory.
// Lane holds row (lane&15); slots map to two contiguous 8-float runs.
__device__ __forceinline__ v16h a_from_f32row32(const float* __restrict__ row, int h) {
  const float4* p0 = (const float4*)(row + 8 * h);        // k = 8h .. 8h+7
  const float4* p1 = (const float4*)(row + 16 + 8 * h);   // k = 16+8h .. 23+8h
  float4 x0 = p0[0], x1 = p0[1];
  float4 y0 = p1[0], y1 = p1[1];
  v16h a;
  a[0] = (_Float16)x0.x; a[1] = (_Float16)x0.y; a[2] = (_Float16)x0.z; a[3] = (_Float16)x0.w;
  a[4] = (_Float16)x1.x; a[5] = (_Float16)x1.y; a[6] = (_Float16)x1.z; a[7] = (_Float16)x1.w;
  a[8] = (_Float16)y0.x; a[9] = (_Float16)y0.y; a[10] = (_Float16)y0.z; a[11] = (_Float16)y0.w;
  a[12] = (_Float16)y1.x; a[13] = (_Float16)y1.y; a[14] = (_Float16)y1.z; a[15] = (_Float16)y1.w;
  return a;
}

// Load one A k-step from a 16x64 f16 LDS activation tile (row stride 64).
__device__ __forceinline__ v16h a_from_lds(const _Float16* __restrict__ tile,
                                           int m, int ks, int h) {
  const v8h* p0 = (const v8h*)(tile + m * 64 + ks * 32 + 8 * h);
  const v8h* p1 = (const v8h*)(tile + m * 64 + ks * 32 + 16 + 8 * h);
  v8h a0 = *p0, a1 = *p1;
  v16h a;
#pragma unroll
  for (int i = 0; i < 8; ++i) { a[i] = a0[i]; a[8 + i] = a1[i]; }
  return a;
}

// Bias + leaky_relu(0.01) epilogue, C/D layout -> f16 LDS tile (stride 64).
template <int CTS>
__device__ __forceinline__ void store_relu(const v8f* acc, _Float16* __restrict__ out,
                                           const float* __restrict__ bias, int lane) {
  const int m = lane & 15, h = lane >> 4;
#pragma unroll
  for (int ct = 0; ct < CTS; ++ct) {
    const float bb = bias[ct * 16 + m];
#pragma unroll
    for (int r = 0; r < 8; ++r) {
      float v = acc[ct][r] + bb;
      v = (v > 0.f) ? v : 0.01f * v;
      out[(r + 8 * h) * 64 + ct * 16 + m] = (_Float16)v;
    }
  }
}

// Hidden layer: 16x64 LDS in -> 16x64 LDS out (K=64 -> 2 ksteps, 4 col tiles).
__device__ __forceinline__ void layer_hidden(const _Float16* __restrict__ in,
                                             _Float16* __restrict__ out,
                                             const v16h* __restrict__ pB,
                                             const float* __restrict__ bias, int lane) {
  const int m = lane & 15, h = lane >> 4;
  v8f acc[4];
#pragma unroll
  for (int ct = 0; ct < 4; ++ct) acc[ct] = zero8();
#pragma unroll
  for (int ks = 0; ks < 2; ++ks) {
    v16h a = a_from_lds(in, m, ks, h);
#pragma unroll
    for (int ct = 0; ct < 4; ++ct)
      acc[ct] = wmma32(a, pB[(ks * 4 + ct) * 32 + lane], acc[ct]);
  }
  store_relu<4>(acc, out, bias, lane);
}

// Output layer: 16x64 LDS in -> f32 accumulators + bias (no activation).
template <int CTS>
__device__ __forceinline__ void layer3_acc(const _Float16* __restrict__ in,
                                           const v16h* __restrict__ pB,
                                           const float* __restrict__ bias, int bias_n,
                                           int lane, float d[][8]) {
  const int m = lane & 15, h = lane >> 4;
  v8f acc[CTS];
#pragma unroll
  for (int ct = 0; ct < CTS; ++ct) acc[ct] = zero8();
#pragma unroll
  for (int ks = 0; ks < 2; ++ks) {
    v16h a = a_from_lds(in, m, ks, h);
#pragma unroll
    for (int ct = 0; ct < CTS; ++ct)
      acc[ct] = wmma32(a, pB[(ks * CTS + ct) * 32 + lane], acc[ct]);
  }
#pragma unroll
  for (int ct = 0; ct < CTS; ++ct) {
    const int col = ct * 16 + m;
    const float bb = (col < bias_n) ? bias[col] : 0.f;
#pragma unroll
    for (int r = 0; r < 8; ++r) d[ct][r] = acc[ct][r] + bb;
  }
}

// LayerNorm over 32 features; row M=r+8h lives in 16 lanes x 2 col tiles.
// xor masks 1,2,4,8 reduce within each 16-lane half (one row group).
__device__ __forceinline__ void layernorm32(float d0[8], float d1[8],
                                            const float* __restrict__ gamma,
                                            const float* __restrict__ beta, int m) {
#pragma unroll
  for (int r = 0; r < 8; ++r) {
    float s = d0[r] + d1[r];
    float q = d0[r] * d0[r] + d1[r] * d1[r];
#pragma unroll
    for (int mask = 1; mask <= 8; mask <<= 1) {
      s += __shfl_xor(s, mask, 32);
      q += __shfl_xor(q, mask, 32);
    }
    float mean = s * (1.f / 32.f);
    float var  = q * (1.f / 32.f) - mean * mean;
    float inv  = rsqrtf(var + 1e-5f);
    d0[r] = (d0[r] - mean) * inv * gamma[m] + beta[m];
    d1[r] = (d1[r] - mean) * inv * gamma[16 + m] + beta[16 + m];
  }
}

// ---------------------------------------------------------------------------
// Weight pre-pack: W[K][N] f32 row-major -> per-(kstep,coltile) register tiles.
// B layout (16-bit, KxN=32x16): lane = col (lane&15), half h holds k=aperm(h,i).
// ---------------------------------------------------------------------------
__global__ void k_pack(const float* __restrict__ W, v16h* __restrict__ out,
                       int K, int N, int cts) {
  const int lane = threadIdx.x;
  const int bid = blockIdx.x;
  const int ks = bid / cts, ct = bid % cts;
  const int m = lane & 15, h = lane >> 4;
  const int col = ct * 16 + m;
  v16h v;
#pragma unroll
  for (int i = 0; i < 16; ++i) {
    int k = ks * 32 + aperm(h, i);
    float x = (k < K && col < N) ? W[(size_t)k * N + col] : 0.f;
    v[i] = (_Float16)x;
  }
  out[(size_t)bid * 32 + lane] = v;
}

__global__ void k_zero(float* __restrict__ p, int n) {
  int i = blockIdx.x * blockDim.x + threadIdx.x;
  if (i < n) p[i] = 0.f;
}

// ---------------------------------------------------------------------------
// Stage kernels: 256 threads = 8 waves; each wave owns 16 graph rows.
// ---------------------------------------------------------------------------

__global__ __launch_bounds__(256) void k_encode_nodes(const float* __restrict__ nfeat,
                                                      float* __restrict__ pn,
                                                      MlpP mp, int nrows) {
  __shared__ __align__(16) _Float16 sh1[8][1024];
  __shared__ __align__(16) _Float16 sh2[8][1024];
  const int wave = threadIdx.x >> 5, lane = threadIdx.x & 31;
  const int m = lane & 15, h = lane >> 4;
  const int base = (blockIdx.x * 8 + wave) * 16;
  if (base >= nrows) return;
  _Float16* h1 = sh1[wave];
  _Float16* h2 = sh2[wave];
  const int row = min(base + m, nrows - 1);

  // layer 1: K=16 (zero-padded to 32)
  v16h a;
  {
    const float4* p = (const float4*)(nfeat + (size_t)row * 16 + 8 * h);
    float4 x0 = p[0], x1 = p[1];
    a[0] = (_Float16)x0.x; a[1] = (_Float16)x0.y; a[2] = (_Float16)x0.z; a[3] = (_Float16)x0.w;
    a[4] = (_Float16)x1.x; a[5] = (_Float16)x1.y; a[6] = (_Float16)x1.z; a[7] = (_Float16)x1.w;
#pragma unroll
    for (int i = 8; i < 16; ++i) a[i] = (_Float16)0.f;   // k>=16 padding
  }
  v8f acc[4];
#pragma unroll
  for (int ct = 0; ct < 4; ++ct) {
    acc[ct] = zero8();
    acc[ct] = wmma32(a, mp.p1[ct * 32 + lane], acc[ct]);
  }
  store_relu<4>(acc, h1, mp.bi, lane);

  layer_hidden(h1, h2, mp.p2, mp.bh, lane);
  float d[2][8];
  layer3_acc<2>(h2, mp.p3, mp.bo, 32, lane, d);
  layernorm32(d[0], d[1], mp.gamma, mp.beta, m);
#pragma unroll
  for (int r = 0; r < 8; ++r) {
    int rr = base + r + 8 * h;
    if (rr < nrows) {
      pn[(size_t)rr * 32 + m] = d[0][r];
      pn[(size_t)rr * 32 + 16 + m] = d[1][r];
    }
  }
}

__global__ __launch_bounds__(256) void k_encode_edges(const float* __restrict__ efeat,
                                                      float* __restrict__ pe,
                                                      MlpP mp, int nrows) {
  __shared__ __align__(16) _Float16 sh1[8][1024];
  __shared__ __align__(16) _Float16 sh2[8][1024];
  const int wave = threadIdx.x >> 5, lane = threadIdx.x & 31;
  const int m = lane & 15, h = lane >> 4;
  const int base = (blockIdx.x * 8 + wave) * 16;
  if (base >= nrows) return;
  _Float16* h1 = sh1[wave];
  _Float16* h2 = sh2[wave];
  const int row = min(base + m, nrows - 1);

  // layer 1: K=5 (zero-padded to 32); only half h=0 slots 0..4 are live.
  v16h a;
#pragma unroll
  for (int i = 0; i < 16; ++i) a[i] = (_Float16)0.f;
  if (h == 0) {
    const float* r5 = efeat + (size_t)row * 5;
#pragma unroll
    for (int i = 0; i < 5; ++i) a[i] = (_Float16)r5[i];
  }
  v8f acc[4];
#pragma unroll
  for (int ct = 0; ct < 4; ++ct) {
    acc[ct] = zero8();
    acc[ct] = wmma32(a, mp.p1[ct * 32 + lane], acc[ct]);
  }
  store_relu<4>(acc, h1, mp.bi, lane);

  layer_hidden(h1, h2, mp.p2, mp.bh, lane);
  float d[2][8];
  layer3_acc<2>(h2, mp.p3, mp.bo, 32, lane, d);
  layernorm32(d[0], d[1], mp.gamma, mp.beta, m);
#pragma unroll
  for (int r = 0; r < 8; ++r) {
    int rr = base + r + 8 * h;
    if (rr < nrows) {
      pe[(size_t)rr * 32 + m] = d[0][r];
      pe[(size_t)rr * 32 + 16 + m] = d[1][r];
    }
  }
}

// Edge update: cat(pe, pn[src], pn[dst]) -> MLP -> LN -> +pe residual;
// fused scatter-add into pe_sum (segment_sum over dst).
__global__ __launch_bounds__(256) void k_edge_update(float* __restrict__ pe,
                                                     const float* __restrict__ pn,
                                                     const int* __restrict__ src,
                                                     const int* __restrict__ dst,
                                                     MlpP mp,
                                                     float* __restrict__ pe_sum,
                                                     int E) {
  __shared__ __align__(16) _Float16 sh1[8][1024];
  __shared__ __align__(16) _Float16 sh2[8][1024];
  const int wave = threadIdx.x >> 5, lane = threadIdx.x & 31;
  const int m = lane & 15, h = lane >> 4;
  const int base = (blockIdx.x * 8 + wave) * 16;
  if (base >= E) return;
  _Float16* h1 = sh1[wave];
  _Float16* h2 = sh2[wave];
  const int e = min(base + m, E - 1);
  const int s = src[e];
  const int t = dst[e];

  // layer 1: K=96 = 3 ksteps: [pe | pn[src] | pn[dst]] without materialized cat
  v8f acc[4];
#pragma unroll
  for (int ct = 0; ct < 4; ++ct) acc[ct] = zero8();
  {
    v16h a = a_from_f32row32(pe + (size_t)e * 32, h);
#pragma unroll
    for (int ct = 0; ct < 4; ++ct)
      acc[ct] = wmma32(a, mp.p1[(0 * 4 + ct) * 32 + lane], acc[ct]);
  }
  {
    v16h a = a_from_f32row32(pn + (size_t)s * 32, h);
#pragma unroll
    for (int ct = 0; ct < 4; ++ct)
      acc[ct] = wmma32(a, mp.p1[(1 * 4 + ct) * 32 + lane], acc[ct]);
  }
  {
    v16h a = a_from_f32row32(pn + (size_t)t * 32, h);
#pragma unroll
    for (int ct = 0; ct < 4; ++ct)
      acc[ct] = wmma32(a, mp.p1[(2 * 4 + ct) * 32 + lane], acc[ct]);
  }
  store_relu<4>(acc, h1, mp.bi, lane);

  layer_hidden(h1, h2, mp.p2, mp.bh, lane);
  float d[2][8];
  layer3_acc<2>(h2, mp.p3, mp.bo, 32, lane, d);
  layernorm32(d[0], d[1], mp.gamma, mp.beta, m);

#pragma unroll
  for (int r = 0; r < 8; ++r) {
    int rr = base + r + 8 * h;
    if (rr < E) {
      float n0 = d[0][r] + pe[(size_t)rr * 32 + m];
      float n1 = d[1][r] + pe[(size_t)rr * 32 + 16 + m];
      pe[(size_t)rr * 32 + m] = n0;
      pe[(size_t)rr * 32 + 16 + m] = n1;
      int dn = dst[rr];
      atomicAdd(&pe_sum[(size_t)dn * 32 + m], n0);
      atomicAdd(&pe_sum[(size_t)dn * 32 + 16 + m], n1);
    }
  }
}

// Node update: cat(pn, pe_sum) -> MLP -> LN -> +pn residual.
__global__ __launch_bounds__(256) void k_node_update(float* __restrict__ pn,
                                                     const float* __restrict__ pe_sum,
                                                     MlpP mp, int N) {
  __shared__ __align__(16) _Float16 sh1[8][1024];
  __shared__ __align__(16) _Float16 sh2[8][1024];
  const int wave = threadIdx.x >> 5, lane = threadIdx.x & 31;
  const int m = lane & 15, h = lane >> 4;
  const int base = (blockIdx.x * 8 + wave) * 16;
  if (base >= N) return;
  _Float16* h1 = sh1[wave];
  _Float16* h2 = sh2[wave];
  const int row = min(base + m, N - 1);

  v8f acc[4];
#pragma unroll
  for (int ct = 0; ct < 4; ++ct) acc[ct] = zero8();
  {
    v16h a = a_from_f32row32(pn + (size_t)row * 32, h);
#pragma unroll
    for (int ct = 0; ct < 4; ++ct)
      acc[ct] = wmma32(a, mp.p1[(0 * 4 + ct) * 32 + lane], acc[ct]);
  }
  {
    v16h a = a_from_f32row32(pe_sum + (size_t)row * 32, h);
#pragma unroll
    for (int ct = 0; ct < 4; ++ct)
      acc[ct] = wmma32(a, mp.p1[(1 * 4 + ct) * 32 + lane], acc[ct]);
  }
  store_relu<4>(acc, h1, mp.bi, lane);

  layer_hidden(h1, h2, mp.p2, mp.bh, lane);
  float d[2][8];
  layer3_acc<2>(h2, mp.p3, mp.bo, 32, lane, d);
  layernorm32(d[0], d[1], mp.gamma, mp.beta, m);

#pragma unroll
  for (int r = 0; r < 8; ++r) {
    int rr = base + r + 8 * h;
    if (rr < N) {
      pn[(size_t)rr * 32 + m]      = d[0][r] + pn[(size_t)rr * 32 + m];
      pn[(size_t)rr * 32 + 16 + m] = d[1][r] + pn[(size_t)rr * 32 + 16 + m];
    }
  }
}

// Decoder: 32 -> 64 -> 64 -> 2 (no LN). Output columns padded to one col tile.
__global__ __launch_bounds__(256) void k_decode(const float* __restrict__ pn,
                                                float* __restrict__ out,
                                                MlpP mp, int N) {
  __shared__ __align__(16) _Float16 sh1[8][1024];
  __shared__ __align__(16) _Float16 sh2[8][1024];
  const int wave = threadIdx.x >> 5, lane = threadIdx.x & 31;
  const int m = lane & 15, h = lane >> 4;
  const int base = (blockIdx.x * 8 + wave) * 16;
  if (base >= N) return;
  _Float16* h1 = sh1[wave];
  _Float16* h2 = sh2[wave];
  const int row = min(base + m, N - 1);

  v8f acc[4];
#pragma unroll
  for (int ct = 0; ct < 4; ++ct) acc[ct] = zero8();
  {
    v16h a = a_from_f32row32(pn + (size_t)row * 32, h);
#pragma unroll
    for (int ct = 0; ct < 4; ++ct)
      acc[ct] = wmma32(a, mp.p1[ct * 32 + lane], acc[ct]);
  }
  store_relu<4>(acc, h1, mp.bi, lane);

  layer_hidden(h1, h2, mp.p2, mp.bh, lane);
  float d[1][8];
  layer3_acc<1>(h2, mp.p3, mp.bo, 2, lane, d);

#pragma unroll
  for (int r = 0; r < 8; ++r) {
    int rr = base + r + 8 * h;
    if (rr < N && m < 2) out[(size_t)rr * 2 + m] = d[0][r];
  }
}

// ---------------------------------------------------------------------------
// Host launcher
// ---------------------------------------------------------------------------
extern "C" void kernel_launch(void* const* d_in, const int* in_sizes, int n_in,
                              void* d_out, int out_size, void* d_ws, size_t ws_size,
                              hipStream_t stream) {
  // Params leaves in setup_inputs() insertion order:
  // enc_n{Wi,bi,Wh0,bh0,Wo,bo,gamma,beta}, enc_e{...}, proc_e[0..2]{...},
  // proc_n[0..2]{...}, dec{Wi,bi,Wh0,bh0,Wo,bo}
  static const int leaf_sizes[70] = {
      16 * 64, 64, 64 * 64, 64, 64 * 32, 32, 32, 32,   // enc_n
      5 * 64,  64, 64 * 64, 64, 64 * 32, 32, 32, 32,   // enc_e
      96 * 64, 64, 64 * 64, 64, 64 * 32, 32, 32, 32,   // proc_e[0]
      96 * 64, 64, 64 * 64, 64, 64 * 32, 32, 32, 32,   // proc_e[1]
      96 * 64, 64, 64 * 64, 64, 64 * 32, 32, 32, 32,   // proc_e[2]
      64 * 64, 64, 64 * 64, 64, 64 * 32, 32, 32, 32,   // proc_n[0]
      64 * 64, 64, 64 * 64, 64, 64 * 32, 32, 32, 32,   // proc_n[1]
      64 * 64, 64, 64 * 64, 64, 64 * 32, 32, 32, 32,   // proc_n[2]
      32 * 64, 64, 64 * 64, 64, 64 * 2,  2,            // dec
  };

  const float* nfeat = (const float*)d_in[0];
  const float* efeat = (const float*)d_in[1];
  const float* leaf[70];
  const int* src;
  const int* dst;
  if (n_in >= 74) {
    // one device buffer per pytree leaf
    for (int i = 0; i < 70; ++i) leaf[i] = (const float*)d_in[2 + i];
    src = (const int*)d_in[n_in - 2];
    dst = (const int*)d_in[n_in - 1];
  } else {
    // params passed as one flat f32 buffer (insertion order)
    const float* p = (const float*)d_in[2];
    size_t off = 0;
    for (int i = 0; i < 70; ++i) { leaf[i] = p + off; off += (size_t)leaf_sizes[i]; }
    src = (const int*)d_in[3];
    dst = (const int*)d_in[4];
  }

  const int N = in_sizes[0] / 16;  // nodes
  const int E = in_sizes[1] / 5;   // edges

  // Workspace carve-up (256B aligned).
  char* ws = (char*)d_ws;
  size_t off = 0;
  auto take = [&](size_t bytes) {
    char* p = ws + off;
    off = (off + bytes + 255) & ~(size_t)255;
    return p;
  };
  float* pn     = (float*)take((size_t)N * 32 * sizeof(float));
  float* pe     = (float*)take((size_t)E * 32 * sizeof(float));
  float* pe_sum = (float*)take((size_t)N * 32 * sizeof(float));

  // MLP descriptors: {leaf base, K1, outn, has_ln}
  struct Desc { int base, K1, outn, ln; };
  const Desc descs[9] = {
      {0, 16, 32, 1},                                   // enc_n
      {8, 5, 32, 1},                                    // enc_e
      {16, 96, 32, 1}, {24, 96, 32, 1}, {32, 96, 32, 1},// proc_e
      {40, 64, 32, 1}, {48, 64, 32, 1}, {56, 64, 32, 1},// proc_n
      {64, 32, 2, 0},                                   // dec
  };

  MlpP mp[9];
  for (int i = 0; i < 9; ++i) {
    const Desc& d = descs[i];
    const int ks1  = (d.K1 + 31) / 32;
    const int cts3 = (d.outn + 15) / 16;
    mp[i].bi    = leaf[d.base + 1];
    mp[i].bh    = leaf[d.base + 3];
    mp[i].bo    = leaf[d.base + 5];
    mp[i].gamma = d.ln ? leaf[d.base + 6] : nullptr;
    mp[i].beta  = d.ln ? leaf[d.base + 7] : nullptr;
    v16h* p1 = (v16h*)take((size_t)ks1 * 4 * 32 * sizeof(v16h));
    v16h* p2 = (v16h*)take((size_t)8 * 32 * sizeof(v16h));
    v16h* p3 = (v16h*)take((size_t)2 * cts3 * 32 * sizeof(v16h));
    mp[i].p1 = p1; mp[i].p2 = p2; mp[i].p3 = p3;
    k_pack<<<ks1 * 4, 32, 0, stream>>>(leaf[d.base + 0], p1, d.K1, 64, 4);
    k_pack<<<8, 32, 0, stream>>>(leaf[d.base + 2], p2, 64, 64, 4);
    k_pack<<<2 * cts3, 32, 0, stream>>>(leaf[d.base + 4], p3, 64, d.outn, cts3);
  }

  const int nblkN = (N + 127) / 128;   // 8 waves x 16 rows per block
  const int nblkE = (E + 127) / 128;

  k_encode_nodes<<<nblkN, 256, 0, stream>>>(nfeat, pn, mp[0], N);
  k_encode_edges<<<nblkE, 256, 0, stream>>>(efeat, pe, mp[1], E);

  for (int it = 0; it < 3; ++it) {
    k_zero<<<(N * 32 + 255) / 256, 256, 0, stream>>>(pe_sum, N * 32);
    k_edge_update<<<nblkE, 256, 0, stream>>>(pe, pn, src, dst, mp[2 + it], pe_sum, E);
    k_node_update<<<nblkN, 256, 0, stream>>>(pn, pe_sum, mp[5 + it], N);
  }

  k_decode<<<nblkN, 256, 0, stream>>>(pn, (float*)d_out, mp[8], N);
}
